// QH2OLlamaAttention_36447092474508
// MI455X (gfx1250) — matrix operand order repaired and checked by reading, not compiled, not run on hardware
//
#include <hip/hip_runtime.h>

// ---------------- CDNA5 WMMA types ----------------
typedef __bf16 bf16_t;
typedef bf16_t bf16x16 __attribute__((ext_vector_type(16)));
typedef bf16_t bf16x8  __attribute__((ext_vector_type(8)));
typedef bf16_t bf16x4  __attribute__((ext_vector_type(4)));
typedef float  f32x8   __attribute__((ext_vector_type(8)));
typedef int    v4i     __attribute__((ext_vector_type(4)));

#define WMMA_BF16(A, B, C) \
  __builtin_amdgcn_wmma_f32_16x16x32_bf16(false, (A), false, (B), (short)0, (C), false, false)

__device__ __forceinline__ bf16x16 cat16(bf16x8 lo, bf16x8 hi) {
  return __builtin_shufflevector(lo, hi, 0, 1, 2, 3, 4, 5, 6, 7, 8, 9, 10, 11, 12, 13, 14, 15);
}

// ---------------- gfx1250 async global->LDS (ASYNCcnt) ----------------
typedef __attribute__((address_space(1))) v4i* gv4i_p;   // global int4*
typedef __attribute__((address_space(3))) v4i* lv4i_p;   // LDS int4*

__device__ __forceinline__ void async_ld_b128(const bf16_t* g, bf16_t* l) {
#if __has_builtin(__builtin_amdgcn_global_load_async_to_lds_b128)
  __builtin_amdgcn_global_load_async_to_lds_b128((gv4i_p)(v4i*)g, (lv4i_p)(v4i*)l, 0, 0);
#else
  asm volatile("global_load_async_to_lds_b128 %0, %1, off"
               :: "v"((unsigned)(unsigned long long)l), "v"(g)
               : "memory");
#endif
}

template <int N>
__device__ __forceinline__ void wait_async() {
#if __has_builtin(__builtin_amdgcn_s_wait_asynccnt)
  __builtin_amdgcn_s_wait_asynccnt(N);
#else
  if constexpr (N == 0) asm volatile("s_wait_asynccnt 0x0" ::: "memory");
  else                  asm volatile("s_wait_asynccnt 0x4" ::: "memory");
#endif
}

// ---------------- problem constants ----------------
constexpr int   S    = 2048;
constexpr int   HID  = 4096;
constexpr int   NH   = 32;
constexpr int   HD   = 128;           // head dim
constexpr int   LIM  = S - 204;       // 1844
constexpr int   NDEF = 143;
constexpr int   NHH  = 266;
constexpr int   CSZ  = 613;
constexpr float QK_SCALE = 0.0883883476483184f;   // 1/sqrt(128)
constexpr float NEG  = -3.0e38f;
constexpr size_t SM  = (size_t)S * HID;           // 8388608
constexpr int   HS   = NH * S;                    // 65536

// =====================================================================
// fp32 -> bf16 flat convert (vectorized: float4 in, 8B out)
// =====================================================================
__global__ void cvt_bf16(const float* __restrict__ x, bf16_t* __restrict__ y, int n) {
  const int i = (blockIdx.x * blockDim.x + threadIdx.x) * 4;
  if (i >= n) return;
  const float4 v = *(const float4*)(x + i);
  bf16x4 o;
  o[0] = (bf16_t)v.x; o[1] = (bf16_t)v.y; o[2] = (bf16_t)v.z; o[3] = (bf16_t)v.w;
  *(bf16x4*)(y + i) = o;
}

// =====================================================================
// fp32 [S][HID] -> bf16 [HID][S] transpose (for V^T attention operand)
// =====================================================================
__global__ __launch_bounds__(256) void transpose_cvt(const float* __restrict__ in,
                                                     bf16_t* __restrict__ out) {
  __shared__ bf16_t tile[32][33];
  const int tx = threadIdx.x & 31;
  const int ty = threadIdx.x >> 5;          // 0..7
  const int s0 = blockIdx.y * 32;
  const int c0 = blockIdx.x * 32;
#pragma unroll
  for (int i = 0; i < 4; ++i)
    tile[ty + i * 8][tx] = (bf16_t)in[(size_t)(s0 + ty + i * 8) * HID + c0 + tx];
  __syncthreads();
#pragma unroll
  for (int i = 0; i < 4; ++i)
    out[(size_t)(c0 + ty + i * 8) * S + s0 + tx] = tile[tx][ty + i * 8];
}

// =====================================================================
// GEMM: C[M,N] = A[M,K] * B[N,K]^T, bf16 in / fp32 out.
// 256 threads = 8 waves in 2(M) x 4(N); C tile 128x128; K tile 32.
// Double-buffered async global->LDS staging (ASYNCcnt), WMMA compute.
// =====================================================================
template <int M, int N, int K>
__global__ __launch_bounds__(256) void gemm_bf16(const bf16_t* __restrict__ A,
                                                 const bf16_t* __restrict__ B,
                                                 float* __restrict__ C) {
  __shared__ bf16_t As[2][128][40];   // pitch 80B -> all 8-elem chunks 16B aligned
  __shared__ bf16_t Bs[2][128][40];
  const int tid  = threadIdx.x;
  const int lane = tid & 31;
  const int half = lane >> 4;
  const int l16  = lane & 15;
  const int wave = tid >> 5;
  const int wm   = wave & 1;        // 0..1
  const int wn   = wave >> 1;       // 0..3
  const int m0   = blockIdx.y * 128;
  const int n0   = blockIdx.x * 128;

  f32x8 acc[4][2];
  f32x8 zero = {};
#pragma unroll
  for (int mt = 0; mt < 4; ++mt)
#pragma unroll
    for (int nt = 0; nt < 2; ++nt) acc[mt][nt] = zero;

  const int r = tid >> 1;
  const int c = (tid & 1) * 16;
  const bf16_t* arow = A + (size_t)(m0 + r) * K + c;
  const bf16_t* brow = B + (size_t)(n0 + r) * K + c;

  constexpr int NKT = K / 32;
  // issue K-tiles 0 and 1 into the two LDS buffers (4 async b128 per thread per tile)
#pragma unroll
  for (int p = 0; p < 2; ++p) {
    async_ld_b128(arow + p * 32,     &As[p][r][c]);
    async_ld_b128(arow + p * 32 + 8, &As[p][r][c + 8]);
    async_ld_b128(brow + p * 32,     &Bs[p][r][c]);
    async_ld_b128(brow + p * 32 + 8, &Bs[p][r][c + 8]);
  }

  for (int it = 0; it < NKT; ++it) {
    const int cur = it & 1;
    if (it + 1 < NKT) wait_async<4>();   // tile `it` resident, `it+1` may fly
    else              wait_async<0>();
    __syncthreads();

    bf16x16 afr[4], bfr[2];
#pragma unroll
    for (int mt = 0; mt < 4; ++mt) {
      const int ar = wm * 64 + mt * 16 + l16;
      afr[mt] = cat16(*(const bf16x8*)&As[cur][ar][half * 8],
                      *(const bf16x8*)&As[cur][ar][16 + half * 8]);
    }
#pragma unroll
    for (int nt = 0; nt < 2; ++nt) {
      const int br = wn * 32 + nt * 16 + l16;
      bfr[nt] = cat16(*(const bf16x8*)&Bs[cur][br][half * 8],
                      *(const bf16x8*)&Bs[cur][br][16 + half * 8]);
    }
#pragma unroll
    for (int mt = 0; mt < 4; ++mt)
#pragma unroll
      for (int nt = 0; nt < 2; ++nt)
        acc[mt][nt] = WMMA_BF16(afr[mt], bfr[nt], acc[mt][nt]);

    __syncthreads();                     // all waves done reading buf `cur`
    if (it + 2 < NKT) {
      const int kb = (it + 2) * 32;
      async_ld_b128(arow + kb,     &As[cur][r][c]);
      async_ld_b128(arow + kb + 8, &As[cur][r][c + 8]);
      async_ld_b128(brow + kb,     &Bs[cur][r][c]);
      async_ld_b128(brow + kb + 8, &Bs[cur][r][c + 8]);
    }
  }
#pragma unroll
  for (int mt = 0; mt < 4; ++mt)
#pragma unroll
    for (int nt = 0; nt < 2; ++nt) {
      const int col = n0 + wn * 32 + nt * 16 + l16;
#pragma unroll
      for (int v = 0; v < 8; ++v) {
        const int row = m0 + wm * 64 + mt * 16 + half * 8 + v;
        C[(size_t)row * N + col] = acc[mt][nt][v];
      }
    }
}

// =====================================================================
// RoPE on q and k in place. layout [S][H*D], position from pos_ids
// =====================================================================
__global__ void rope_kernel(float* __restrict__ q, float* __restrict__ k,
                            const int* __restrict__ pos_ids) {
  const int idx = blockIdx.x * blockDim.x + threadIdx.x;  // over S*NH*64
  if (idx >= S * NH * (HD / 2)) return;
  const int d  = idx & 63;
  const int hs = idx >> 6;
  const int h  = hs & (NH - 1);
  const int s  = hs / NH;
  const float inv = __powf(10000.0f, -2.0f * (float)d / (float)HD);
  const float fr  = (float)pos_ids[s] * inv;
  float sn, cs;
  __sincosf(fr, &sn, &cs);
  const size_t base = (size_t)s * HID + h * HD;
  const float q1 = q[base + d], q2 = q[base + d + 64];
  q[base + d]      = q1 * cs - q2 * sn;
  q[base + d + 64] = q2 * cs + q1 * sn;
  const float k1 = k[base + d], k2 = k[base + d + 64];
  k[base + d]      = k1 * cs - k2 * sn;
  k[base + d + 64] = k2 * cs + k1 * sn;
}

// =====================================================================
// Per-(s,h)-row int4 fake quant + L2 error.  1 wave / row, 8 rows / block.
// =====================================================================
__global__ __launch_bounds__(256) void quant_kernel(const float* __restrict__ x,
                                                    float* __restrict__ xq,
                                                    float* __restrict__ err,
                                                    int nbit) {
  const int wave = threadIdx.x >> 5;
  const int lane = threadIdx.x & 31;
  const int row  = blockIdx.x * 8 + wave;      // row index over [S][NH]
  if (row >= S * NH) return;
  const int s = row / NH, h = row % NH;
  const float* p = x + (size_t)row * HD;
  float vals[4];
  float mx = -3.4e38f, mn = 3.4e38f;
#pragma unroll
  for (int i = 0; i < 4; ++i) {
    vals[i] = p[lane + 32 * i];
    mx = fmaxf(mx, vals[i]);
    mn = fminf(mn, vals[i]);
  }
#pragma unroll
  for (int off = 16; off; off >>= 1) {
    mx = fmaxf(mx, __shfl_xor(mx, off, 32));
    mn = fminf(mn, __shfl_xor(mn, off, 32));
  }
  const float maxint = (float)((1 << nbit) - 1);
  const float scale  = fmaxf(mx - mn, 1e-5f) / maxint;
  const float zero   = fminf(fmaxf(rintf(-mn / scale), 0.f), maxint);
  float* o = xq + (size_t)row * HD;
  float e2 = 0.f;
#pragma unroll
  for (int i = 0; i < 4; ++i) {
    const float qv = (fminf(fmaxf(rintf(vals[i] / scale) + zero, 0.f), maxint) - zero) * scale;
    o[lane + 32 * i] = qv;
    const float d = qv - vals[i];
    e2 += d * d;
  }
#pragma unroll
  for (int off = 16; off; off >>= 1) e2 += __shfl_xor(e2, off, 32);
  if (lane == 0) err[h * S + s] = sqrtf(e2);
}

// =====================================================================
// Flash attention pass 1: one wave per (head, 16-query tile), bf16 inputs.
// QK^T and P*V on WMMA, online softmax, saves m/l.
// =====================================================================
__global__ __launch_bounds__(32) void attn_pass1(const bf16_t* __restrict__ qb,
                                                 const bf16_t* __restrict__ kqb,
                                                 const bf16_t* __restrict__ vqT,
                                                 float* __restrict__ ob,
                                                 float* __restrict__ marr,
                                                 float* __restrict__ larr) {
  __shared__ bf16_t Pst[16][40];
  const int lane  = threadIdx.x & 31;
  const int half  = lane >> 4;
  const int l16   = lane & 15;
  const int head  = blockIdx.y;
  const int qbase = blockIdx.x * 16;

  bf16x16 qf[4];
  const bf16_t* qrow = qb + (size_t)(qbase + l16) * HID + head * HD;
#pragma unroll
  for (int kc = 0; kc < 4; ++kc)
    qf[kc] = cat16(*(const bf16x8*)(qrow + kc * 32 + half * 8),
                   *(const bf16x8*)(qrow + kc * 32 + 16 + half * 8));

  f32x8 oacc[8];
  f32x8 zero = {};
#pragma unroll
  for (int t = 0; t < 8; ++t) oacc[t] = zero;
  float m_r[8], l_r[8];
#pragma unroll
  for (int v = 0; v < 8; ++v) { m_r[v] = NEG; l_r[v] = 0.f; }

  for (int kb = 0; kb < qbase + 16; kb += 32) {
    f32x8 s0 = zero, s1 = zero;
#pragma unroll
    for (int kc = 0; kc < 4; ++kc) {
      const bf16_t* kr0 = kqb + (size_t)(kb + l16) * HID + head * HD + kc * 32 + half * 8;
      const bf16_t* kr1 = kqb + (size_t)(kb + 16 + l16) * HID + head * HD + kc * 32 + half * 8;
      const bf16x16 b0 = cat16(*(const bf16x8*)kr0, *(const bf16x8*)(kr0 + 16));
      const bf16x16 b1 = cat16(*(const bf16x8*)kr1, *(const bf16x8*)(kr1 + 16));
      s0 = WMMA_BF16(qf[kc], b0, s0);
      s1 = WMMA_BF16(qf[kc], b1, s1);
    }
    float fmul[8];
#pragma unroll
    for (int v = 0; v < 8; ++v) {
      const int qi = qbase + half * 8 + v;
      const float a0 = (kb + l16      <= qi) ? s0[v] * QK_SCALE : NEG;
      const float a1 = (kb + 16 + l16 <= qi) ? s1[v] * QK_SCALE : NEG;
      float tmax = fmaxf(a0, a1);
#pragma unroll
      for (int off = 8; off; off >>= 1) tmax = fmaxf(tmax, __shfl_xor(tmax, off, 32));
      const float mnew = fmaxf(m_r[v], tmax);
      const float p0 = __expf(a0 - mnew);
      const float p1 = __expf(a1 - mnew);
      float rs = p0 + p1;
#pragma unroll
      for (int off = 8; off; off >>= 1) rs += __shfl_xor(rs, off, 32);
      const float f = __expf(m_r[v] - mnew);
      l_r[v] = l_r[v] * f + rs;
      m_r[v] = mnew;
      fmul[v] = f;
      Pst[half * 8 + v][l16]      = (bf16_t)p0;
      Pst[half * 8 + v][16 + l16] = (bf16_t)p1;
    }
#pragma unroll
    for (int t = 0; t < 8; ++t)
#pragma unroll
      for (int v = 0; v < 8; ++v) oacc[t][v] *= fmul[v];
    __syncthreads();
    const bf16x16 pf = cat16(*(const bf16x8*)&Pst[l16][half * 8],
                             *(const bf16x8*)&Pst[l16][16 + half * 8]);
#pragma unroll
    for (int t = 0; t < 8; ++t) {
      const bf16_t* vp = vqT + (size_t)(head * HD + t * 16 + l16) * S + kb + half * 8;
      const bf16x16 vf = cat16(*(const bf16x8*)vp, *(const bf16x8*)(vp + 16));
      oacc[t] = WMMA_BF16(pf, vf, oacc[t]);
    }
    __syncthreads();
  }
#pragma unroll
  for (int v = 0; v < 8; ++v) {
    const int row = qbase + half * 8 + v;
    const float invl = 1.0f / l_r[v];
#pragma unroll
    for (int t = 0; t < 8; ++t)
      ob[(size_t)row * HID + head * HD + t * 16 + l16] = oacc[t][v] * invl;
    if (l16 == 0) { marr[head * S + row] = m_r[v]; larr[head * S + row] = l_r[v]; }
  }
}

// =====================================================================
// Pass 2: recompute score tiles (WMMA), hh[h,k] += sum_q probs[q,k]
// =====================================================================
__global__ __launch_bounds__(32) void attn_pass2(const bf16_t* __restrict__ qb,
                                                 const bf16_t* __restrict__ kqb,
                                                 const float* __restrict__ marr,
                                                 const float* __restrict__ larr,
                                                 float* __restrict__ hh) {
  const int lane  = threadIdx.x & 31;
  const int half  = lane >> 4;
  const int l16   = lane & 15;
  const int head  = blockIdx.y;
  const int qbase = blockIdx.x * 16;

  bf16x16 qf[4];
  const bf16_t* qrow = qb + (size_t)(qbase + l16) * HID + head * HD;
#pragma unroll
  for (int kc = 0; kc < 4; ++kc)
    qf[kc] = cat16(*(const bf16x8*)(qrow + kc * 32 + half * 8),
                   *(const bf16x8*)(qrow + kc * 32 + 16 + half * 8));

  float m_r[8], il_r[8];
#pragma unroll
  for (int v = 0; v < 8; ++v) {
    const int row = qbase + half * 8 + v;
    m_r[v]  = marr[head * S + row];
    il_r[v] = 1.0f / larr[head * S + row];
  }
  f32x8 zero = {};
  for (int kb = 0; kb < qbase + 16; kb += 32) {
    f32x8 s0 = zero, s1 = zero;
#pragma unroll
    for (int kc = 0; kc < 4; ++kc) {
      const bf16_t* kr0 = kqb + (size_t)(kb + l16) * HID + head * HD + kc * 32 + half * 8;
      const bf16_t* kr1 = kqb + (size_t)(kb + 16 + l16) * HID + head * HD + kc * 32 + half * 8;
      const bf16x16 b0 = cat16(*(const bf16x8*)kr0, *(const bf16x8*)(kr0 + 16));
      const bf16x16 b1 = cat16(*(const bf16x8*)kr1, *(const bf16x8*)(kr1 + 16));
      s0 = WMMA_BF16(qf[kc], b0, s0);
      s1 = WMMA_BF16(qf[kc], b1, s1);
    }
    float c0 = 0.f, c1 = 0.f;
#pragma unroll
    for (int v = 0; v < 8; ++v) {
      const int qi = qbase + half * 8 + v;
      const float a0 = (kb + l16      <= qi) ? s0[v] * QK_SCALE : NEG;
      const float a1 = (kb + 16 + l16 <= qi) ? s1[v] * QK_SCALE : NEG;
      c0 += __expf(a0 - m_r[v]) * il_r[v];
      c1 += __expf(a1 - m_r[v]) * il_r[v];
    }
    c0 += __shfl_xor(c0, 16, 32);   // combine the two row halves
    c1 += __shfl_xor(c1, 16, 32);
    if (half == 0) atomicAdd(&hh[head * S + kb + l16], c0);
    else           atomicAdd(&hh[head * S + kb + 16 + l16], c1);
  }
}

// =====================================================================
// small helpers
// =====================================================================
__global__ void init_scalars(unsigned* sc) {
  if (threadIdx.x < 3) {
    sc[threadIdx.x * 2]     = 0x7F7FFFFFu;  // min slots = +FLT_MAX
    sc[threadIdx.x * 2 + 1] = 0u;           // max slots = 0 (data nonneg)
  }
}

__global__ void minmax_kernel(const float* __restrict__ x, int n,
                              unsigned* mn, unsigned* mx) {
  float lmin = 3.4e38f, lmax = 0.0f;   // inputs nonneg -> uint ordering valid
  for (int i = blockIdx.x * blockDim.x + threadIdx.x; i < n; i += gridDim.x * blockDim.x) {
    const float v = x[i];
    lmin = fminf(lmin, v);
    lmax = fmaxf(lmax, v);
  }
#pragma unroll
  for (int off = 16; off; off >>= 1) {
    lmin = fminf(lmin, __shfl_xor(lmin, off, 32));
    lmax = fmaxf(lmax, __shfl_xor(lmax, off, 32));
  }
  if ((threadIdx.x & 31) == 0) {
    atomicMin(mn, __float_as_uint(lmin));
    atomicMax(mx, __float_as_uint(lmax));
  }
}

__global__ void comb_kernel(const float* __restrict__ hh, const float* __restrict__ ek,
                            const float* __restrict__ ev, const unsigned* __restrict__ sc,
                            float* __restrict__ comb) {
  const int i = blockIdx.x * blockDim.x + threadIdx.x;
  if (i >= HS) return;
  const float ekn = (ek[i] - __uint_as_float(sc[0])) / (__uint_as_float(sc[1]) - __uint_as_float(sc[0]));
  const float evn = (ev[i] - __uint_as_float(sc[2])) / (__uint_as_float(sc[3]) - __uint_as_float(sc[2]));
  const float hhn = (hh[i] - __uint_as_float(sc[4])) / (__uint_as_float(sc[5]) - __uint_as_float(sc[4]));
  const float qs  = (2.0f - ekn - evn) * 0.5f;
  comb[i] = 0.5f * hhn + 0.5f * qs;
}

// per-head top-k via rank counting (exact jax top_k tie semantics)
__global__ __launch_bounds__(256) void topk_kernel(const float* __restrict__ hh,
                                                   const float* __restrict__ comb,
                                                   int* __restrict__ keepmask) {
  __shared__ float vals[LIM];
  __shared__ unsigned char defm[LIM];
  const int h = blockIdx.x;
  for (int s2 = threadIdx.x; s2 < LIM; s2 += 256) vals[s2] = hh[h * S + s2];
  __syncthreads();
  for (int s2 = threadIdx.x; s2 < LIM; s2 += 256) {
    const float v = vals[s2];
    int rank = 0;
    for (int j = 0; j < LIM; ++j) {
      const float w = vals[j];
      rank += (int)(w > v) | ((int)(w == v) & (int)(j < s2));
    }
    defm[s2] = (rank < NDEF) ? 1 : 0;
  }
  __syncthreads();
  for (int s2 = threadIdx.x; s2 < LIM; s2 += 256)
    vals[s2] = defm[s2] ? -1.0f : comb[h * S + s2];
  __syncthreads();
  for (int s2 = threadIdx.x; s2 < LIM; s2 += 256) {
    const float v = vals[s2];
    int rank = 0;
    for (int j = 0; j < LIM; ++j) {
      const float w = vals[j];
      rank += (int)(w > v) | ((int)(w == v) & (int)(j < s2));
    }
    keepmask[h * S + s2] = (int)defm[s2] | (int)(rank < NHH);
  }
  for (int s2 = LIM + threadIdx.x; s2 < S; s2 += 256) keepmask[h * S + s2] = 1;
}

__global__ __launch_bounds__(256) void compact_kernel(const int* __restrict__ keepmask,
                                                      int* __restrict__ keepidx) {
  __shared__ int pc[256];
  const int h = blockIdx.x;
  const int base_s = threadIdx.x * 8;   // S/256 == 8
  int cnt = 0;
#pragma unroll
  for (int i = 0; i < 8; ++i) cnt += keepmask[h * S + base_s + i];
  pc[threadIdx.x] = cnt;
  __syncthreads();
  if (threadIdx.x == 0) {
    int run = 0;
    for (int i = 0; i < 256; ++i) { const int c = pc[i]; pc[i] = run; run += c; }
  }
  __syncthreads();
  int pos = pc[threadIdx.x];
#pragma unroll
  for (int i = 0; i < 8; ++i)
    if (keepmask[h * S + base_s + i]) keepidx[h * CSZ + pos++] = base_s + i;
}

__global__ void gather_kernel(const float* __restrict__ kq, const float* __restrict__ vq,
                              const int* __restrict__ keepidx,
                              float* __restrict__ khh, float* __restrict__ vhh) {
  const int h = blockIdx.y, pos = blockIdx.x, d = threadIdx.x * 4;  // 32 thr * float4
  const int s = keepidx[h * CSZ + pos];
  const size_t src = (size_t)s * HID + h * HD + d;
  const size_t dst = ((size_t)h * CSZ + pos) * HD + d;
  *(float4*)(khh + dst) = *(const float4*)(kq + src);
  *(float4*)(vhh + dst) = *(const float4*)(vq + src);
}

// =====================================================================
// launch
// =====================================================================
extern "C" void kernel_launch(void* const* d_in, const int* in_sizes, int n_in,
                              void* d_out, int out_size, void* d_ws, size_t ws_size,
                              hipStream_t stream) {
  (void)in_sizes; (void)n_in; (void)out_size; (void)ws_size;
  const float* hid = (const float*)d_in[0];
  const float* wq  = (const float*)d_in[1];
  const float* wk  = (const float*)d_in[2];
  const float* wv  = (const float*)d_in[3];
  const float* wo  = (const float*)d_in[4];
  const int*   pid = (const int*)d_in[5];

  float* ws    = (float*)d_ws;
  float* qbuf  = ws;               // fp32 activations
  float* kbuf  = ws + 1 * SM;
  float* vbuf  = ws + 2 * SM;
  float* kqbuf = ws + 3 * SM;
  float* vqbuf = ws + 4 * SM;
  float* abuf  = ws + 5 * SM;
  float* ek    = ws + 6 * SM;      // small fp32 arrays
  float* ev    = ek + HS;
  float* hh    = ev + HS;
  float* comb  = hh + HS;
  float* marr  = comb + HS;
  float* larr  = marr + HS;
  int*   keepmask = (int*)(larr + HS);
  int*   keepidx  = keepmask + HS;
  unsigned* sc    = (unsigned*)(keepidx + NH * CSZ);
  // bf16 region (16B-aligned: all offsets are multiples of 4 floats)
  bf16_t* hidbf = (bf16_t*)(ws + 6 * SM + 8 * (size_t)HS);
  bf16_t* wbf   = hidbf + SM;          // HID*HID = 2*SM elems, reused per weight
  bf16_t* qbf   = wbf + 2 * SM;
  bf16_t* kqbf  = qbf + SM;
  bf16_t* vqTbf = kqbf + SM;
  bf16_t* abf   = vqTbf + SM;

  float* outp = (float*)d_out;
  float* khh  = outp + SM;
  float* vhh  = khh + (size_t)NH * CSZ * HD;

  const int W2 = HID * HID;                       // weight element count
  const dim3 ggrid(HID / 128, S / 128);           // (32,16)
  const dim3 tgrid(HID / 32, S / 32);             // transpose grid

  // ---- projections: convert once to bf16, then WMMA GEMMs ----
  cvt_bf16<<<(int)(SM / 4 + 255) / 256, 256, 0, stream>>>(hid, hidbf, (int)SM);
  cvt_bf16<<<(W2 / 4 + 255) / 256, 256, 0, stream>>>(wq, wbf, W2);
  gemm_bf16<S, HID, HID><<<ggrid, 256, 0, stream>>>(hidbf, wbf, qbuf);
  cvt_bf16<<<(W2 / 4 + 255) / 256, 256, 0, stream>>>(wk, wbf, W2);
  gemm_bf16<S, HID, HID><<<ggrid, 256, 0, stream>>>(hidbf, wbf, kbuf);
  cvt_bf16<<<(W2 / 4 + 255) / 256, 256, 0, stream>>>(wv, wbf, W2);
  gemm_bf16<S, HID, HID><<<ggrid, 256, 0, stream>>>(hidbf, wbf, vbuf);

  rope_kernel<<<(S * NH * 64 + 255) / 256, 256, 0, stream>>>(qbuf, kbuf, pid);

  quant_kernel<<<(S * NH) / 8, 256, 0, stream>>>(kbuf, kqbuf, ek, 4);
  quant_kernel<<<(S * NH) / 8, 256, 0, stream>>>(vbuf, vqbuf, ev, 4);

  // ---- bf16 attention operands ----
  cvt_bf16<<<(int)(SM / 4 + 255) / 256, 256, 0, stream>>>(qbuf, qbf, (int)SM);
  cvt_bf16<<<(int)(SM / 4 + 255) / 256, 256, 0, stream>>>(kqbuf, kqbf, (int)SM);
  transpose_cvt<<<tgrid, 256, 0, stream>>>(vqbuf, vqTbf);

  init_scalars<<<1, 32, 0, stream>>>(sc);
  (void)hipMemsetAsync(hh, 0, HS * sizeof(float), stream);

  attn_pass1<<<dim3(S / 16, NH), 32, 0, stream>>>(qbf, kqbf, vqTbf, abuf, marr, larr);
  attn_pass2<<<dim3(S / 16, NH), 32, 0, stream>>>(qbf, kqbf, marr, larr, hh);

  // ---- output projection ----
  cvt_bf16<<<(int)(SM / 4 + 255) / 256, 256, 0, stream>>>(abuf, abf, (int)SM);
  cvt_bf16<<<(W2 / 4 + 255) / 256, 256, 0, stream>>>(wo, wbf, W2);
  gemm_bf16<S, HID, HID><<<ggrid, 256, 0, stream>>>(abf, wbf, outp);

  // ---- H2O selection ----
  minmax_kernel<<<64, 256, 0, stream>>>(ek, HS, sc + 0, sc + 1);
  minmax_kernel<<<64, 256, 0, stream>>>(ev, HS, sc + 2, sc + 3);
  minmax_kernel<<<64, 256, 0, stream>>>(hh, HS, sc + 4, sc + 5);

  comb_kernel<<<(HS + 255) / 256, 256, 0, stream>>>(hh, ek, ev, sc, comb);
  topk_kernel<<<NH, 256, 0, stream>>>(hh, comb, keepmask);
  compact_kernel<<<NH, 256, 0, stream>>>(keepmask, keepidx);
  gather_kernel<<<dim3(CSZ, NH), 32, 0, stream>>>(kqbuf, vqbuf, keepidx, khh, vhh);
}
